// GraphAttentionLayer_87686052315947
// MI455X (gfx1250) — matrix operand order repaired and checked by reading, not compiled, not run on hardware
//
#include <hip/hip_runtime.h>
#include <hip/hip_bf16.h>

// GAT layer on MI455X (gfx1250, wave32, WMMA + TDM).
// K1: Wh = h@W via bf16 WMMA, stored TRANSPOSED (WhT[b][n][j]) + s1,s2.
// K2: single-pass flash attention; Wh value tiles staged to LDS by the
//     Tensor Data Mover (tensor_load_to_lds), double-buffered against WMMA.

typedef __attribute__((ext_vector_type(16))) __bf16 v16bf;
typedef __attribute__((ext_vector_type(8)))  float  v8f;
typedef __attribute__((ext_vector_type(4)))  unsigned int v4u;
typedef __attribute__((ext_vector_type(8)))  int    v8i;
typedef __attribute__((ext_vector_type(4)))  int    v4i;

constexpr int Bsz = 8;     // batch
constexpr int N   = 2048;  // nodes
constexpr int FI  = 256;   // F_in
constexpr int FO  = 128;   // F_out
constexpr float ALPHA = 0.2f;

__device__ __forceinline__ float lrelu(float x) { return x > 0.0f ? x : ALPHA * x; }

// ---------------------------------------------------------------------------
// TDM: 2D tile load Global -> LDS. Descriptor bit layout per ISA 08 §8.3-8.6.
//   tile_d0 contiguous elements (2B each) per row, tile_d1 rows,
//   row stride = stride_e elements.
// ---------------------------------------------------------------------------
__device__ __forceinline__ void tdm_load_2d(unsigned long long gaddr, unsigned int lds_off,
                                            unsigned int tensor_d0, unsigned int tensor_d1,
                                            unsigned int tile_d0, unsigned int tile_d1,
                                            unsigned int stride_e)
{
    v4u g0 = { 1u,                                    // count=1, no gather/restore
               lds_off,                               // lds_addr (byte offset)
               (unsigned int)gaddr,                   // global_addr[31:0]
               (unsigned int)((gaddr >> 32) & 0x01FFFFFFu) | (2u << 30) }; // [56:32] | type=2
    v8i g1 = { (int)(1u << 16),                                   // data_size=1 (2 bytes)
               (int)((tensor_d0 & 0xFFFFu) << 16),                // tensor_dim0[15:0]
               (int)(((tensor_d0 >> 16) & 0xFFFFu) |
                     ((tensor_d1 & 0xFFFFu) << 16)),              // dim0 hi | dim1 lo
               (int)(((tensor_d1 >> 16) & 0xFFFFu) |
                     ((tile_d0 & 0xFFFFu) << 16)),                // dim1 hi | tile_dim0
               (int)(tile_d1 & 0xFFFFu),                          // tile_dim1 (tile_dim2=0)
               (int)stride_e,                                     // tensor_dim0_stride lo32
               0,                                                 // stride hi | dim1_stride lo
               0 };
    v4i g2 = { 0, 0, 0, 0 };
    v4i g3 = { 0, 0, 0, 0 };
#if defined(__clang_major__) && (__clang_major__ >= 23)
    v8i g4 = { 0, 0, 0, 0, 0, 0, 0, 0 };
    __builtin_amdgcn_tensor_load_to_lds(g0, g1, g2, g3, g4, 0);
#else
    __builtin_amdgcn_tensor_load_to_lds(g0, g1, g2, g3, 0);
#endif
}

__device__ __forceinline__ unsigned int lds_offset_of(const void* p)
{
    // Generic pointers into LDS carry the LDS byte offset in addr[31:0]
    // (ISA 00 §10.2 aperture mapping).
    return (unsigned int)(unsigned long long)(uintptr_t)p;
}

// ---------------------------------------------------------------------------
// Kernel 1: Wh = h @ W (bf16 WMMA, f32 accum) -> WhT (transposed, bf16), s1, s2.
// ---------------------------------------------------------------------------
__global__ void __launch_bounds__(128)
gat_gemm_scores(const float* __restrict__ h, const float* __restrict__ W,
                const float* __restrict__ a, __bf16* __restrict__ WhT,
                float* __restrict__ s1, float* __restrict__ s2)
{
    __shared__ __bf16 Wt[FO * FI];  // 64 KB: Wt[n*FI + k] = bf16(W[k][n])

    for (int idx = threadIdx.x; idx < FI * FO; idx += 128) {
        int k = idx >> 7, n = idx & 127;
        Wt[n * FI + k] = (__bf16)W[idx];
    }
    __syncthreads();

    const int b    = blockIdx.x / (N / 64);
    const int row0 = (blockIdx.x % (N / 64)) * 64;
    const int w    = threadIdx.x >> 5;
    const int lane = threadIdx.x & 31;
    const int hi   = (lane >= 16) ? 1 : 0;
    const int li   = lane & 15;

    const int rowA = row0 + w * 16 + li;              // A-fragment row (M = lane&15)
    const float* hrow = h + ((size_t)(b * N + rowA)) * FI;

    v8f acc[8] = {};

    for (int kt = 0; kt < 8; ++kt) {
        const int k0  = kt * 32;
        const int klo = k0 + (hi ? 16 : 0);
        const int khi = k0 + (hi ? 24 : 8);
        // A fragment: 16x32 bf16, per-lane K contiguous (k = 16*hi + s), same as B
        float4 f0 = *(const float4*)(hrow + klo);
        float4 f1 = *(const float4*)(hrow + klo + 4);
        float4 f2 = *(const float4*)(hrow + khi);
        float4 f3 = *(const float4*)(hrow + khi + 4);
        v16bf af;
        af[0] = (__bf16)f0.x;  af[1] = (__bf16)f0.y;  af[2]  = (__bf16)f0.z;  af[3]  = (__bf16)f0.w;
        af[4] = (__bf16)f1.x;  af[5] = (__bf16)f1.y;  af[6]  = (__bf16)f1.z;  af[7]  = (__bf16)f1.w;
        af[8] = (__bf16)f2.x;  af[9] = (__bf16)f2.y;  af[10] = (__bf16)f2.z;  af[11] = (__bf16)f2.w;
        af[12]= (__bf16)f3.x;  af[13]= (__bf16)f3.y;  af[14] = (__bf16)f3.z;  af[15] = (__bf16)f3.w;

#pragma unroll
        for (int nt = 0; nt < 8; ++nt) {
            const int n = nt * 16 + li;               // B col (N = lane&15)
            v16bf bf_ = *(const v16bf*)(Wt + n * FI + k0 + (hi ? 16 : 0));
            acc[nt] = __builtin_amdgcn_wmma_f32_16x16x32_bf16(
                false, af, false, bf_, (short)0, acc[nt], false, false);
        }
    }

    // Epilogue: store WhT (bf16, feature-major) + per-row dots with a1/a2.
    // C layout: lane holds rows m = r + 8*hi, col n = nt*16 + li. In WhT the
    // 8 row-values of one lane are 8 consecutive bf16 -> one uint4 store.
    float pr1[8] = {}, pr2[8] = {};
#pragma unroll
    for (int nt = 0; nt < 8; ++nt) {
        const float a1v = a[nt * 16 + li];
        const float a2v = a[FO + nt * 16 + li];
        union { uint4 u; __bf16 hh[8]; } pk;
#pragma unroll
        for (int r = 0; r < 8; ++r) {
            const float v = acc[nt][r];
            pk.hh[r] = (__bf16)v;
            pr1[r] += v * a1v;
            pr2[r] += v * a2v;
        }
        *(uint4*)(WhT + (size_t)(b * FO + nt * 16 + li) * N
                      + row0 + w * 16 + (hi ? 8 : 0)) = pk.u;
    }
#pragma unroll
    for (int r = 0; r < 8; ++r) {
        float v1 = pr1[r], v2 = pr2[r];
#pragma unroll
        for (int msk = 1; msk <= 8; msk <<= 1) {      // reduce over 16 lanes of half-wave
            v1 += __shfl_xor(v1, msk, 32);
            v2 += __shfl_xor(v2, msk, 32);
        }
        if (li == 0) {
            const int g = b * N + row0 + w * 16 + r + (hi ? 8 : 0);
            s1[g] = v1;
            s2[g] = v2;
        }
    }
}

// ---------------------------------------------------------------------------
// Kernel 1b: per-batch max of s2 (softmax row max; exact since LR monotonic).
// ---------------------------------------------------------------------------
__global__ void __launch_bounds__(256)
gat_s2max(const float* __restrict__ s2, float* __restrict__ s2max)
{
    __shared__ float red[256];
    float m = -3.4e38f;
    for (int idx = threadIdx.x; idx < N; idx += 256)
        m = fmaxf(m, s2[blockIdx.x * N + idx]);
    red[threadIdx.x] = m;
    __syncthreads();
    for (int s = 128; s > 0; s >>= 1) {
        if (threadIdx.x < s) red[threadIdx.x] = fmaxf(red[threadIdx.x], red[threadIdx.x + s]);
        __syncthreads();
    }
    if (threadIdx.x == 0) s2max[blockIdx.x] = red[0];
}

// ---------------------------------------------------------------------------
// Kernel 2: out = elu(softmax(LR(s1_i+s2_j)) @ Wh), single pass.
// Value tiles (128 features x 32 j, bf16) staged LDS<-global by TDM,
// double-buffered: DMA of tile t+1 overlaps WMMA on tile t.
// ---------------------------------------------------------------------------
__global__ void __launch_bounds__(128)
gat_attention(const __bf16* __restrict__ WhT, const float* __restrict__ s1,
              const float* __restrict__ s2, const float* __restrict__ s2max,
              float* __restrict__ out)
{
    __shared__ float  s2sh[N];           // 8 KB
    __shared__ __bf16 Vt[2][FO * 32];    // 2 x 8 KB: Vt[buf][n*32 + j]

    const int b    = blockIdx.x / (N / 64);
    const int row0 = (blockIdx.x % (N / 64)) * 64;
    const int w    = threadIdx.x >> 5;
    const int lane = threadIdx.x & 31;
    const int hi   = (lane >= 16) ? 1 : 0;
    const int li   = lane & 15;

    const __bf16* vbase = WhT + (size_t)b * FO * N;

    // Kick off DMA of tile 0 while we stage s2.
    if (w == 0)
        tdm_load_2d((unsigned long long)(uintptr_t)vbase,
                    lds_offset_of(&Vt[0][0]), N, FO, 32, FO, N);

    for (int idx = threadIdx.x; idx < N; idx += 128)
        s2sh[idx] = s2[b * N + idx];

    const int rowA = row0 + w * 16 + li;
    const float s1i = s1[b * N + rowA];
    const float mi  = lrelu(s1i + s2max[b]);

    float lpart = 0.0f;
    v8f acc[8] = {};

    constexpr int T = N / 32;            // 64 j-tiles
    for (int t = 0; t < T; ++t) {
        const int buf = t & 1;
        if (w == 0)
            __builtin_amdgcn_s_wait_tensorcnt(0);   // tile t resident in Vt[buf]
        __syncthreads();                            // publish; all done reading Vt[buf^1]
        if (w == 0 && t + 1 < T)
            tdm_load_2d((unsigned long long)(uintptr_t)(vbase + (t + 1) * 32),
                        lds_offset_of(&Vt[buf ^ 1][0]), N, FO, 32, FO, N);

        const int j0 = t * 32;
        // A fragment = P tile (16 rows x 32 cols), built in registers.
        const int klo = hi ? 16 : 0, khi = hi ? 24 : 8;
        v16bf af;
#pragma unroll
        for (int e = 0; e < 8; ++e) {
            float p = __expf(lrelu(s1i + s2sh[j0 + klo + e]) - mi);
            lpart += p;
            af[e] = (__bf16)p;
            float q = __expf(lrelu(s1i + s2sh[j0 + khi + e]) - mi);
            lpart += q;
            af[e + 8] = (__bf16)q;
        }

#pragma unroll
        for (int nt = 0; nt < 8; ++nt) {
            const int n = nt * 16 + li;
            v16bf bf_ = *(const v16bf*)(&Vt[buf][0] + n * 32 + (hi ? 16 : 0));
            acc[nt] = __builtin_amdgcn_wmma_f32_16x16x32_bf16(
                false, af, false, bf_, (short)0, acc[nt], false, false);
        }
    }

    // l_i: lanes i and i+16 both hold partials for row (lane&15)
    float lfull = lpart + __shfl_xor(lpart, 16, 32);
    const float inv = 1.0f / lfull;
    float invl[8];
#pragma unroll
    for (int r = 0; r < 8; ++r)
        invl[r] = __shfl(inv, r + (hi ? 8 : 0), 32);  // row m = r+8*hi lives at lane m

#pragma unroll
    for (int nt = 0; nt < 8; ++nt) {
#pragma unroll
        for (int r = 0; r < 8; ++r) {
            const int m = r + (hi ? 8 : 0);
            float v = acc[nt][r] * invl[r];
            v = v > 0.0f ? v : __expf(v) - 1.0f;      // ELU
            out[(size_t)(b * N + row0 + w * 16 + m) * FO + nt * 16 + li] = v;
        }
    }
}

// ---------------------------------------------------------------------------
extern "C" void kernel_launch(void* const* d_in, const int* in_sizes, int n_in,
                              void* d_out, int out_size, void* d_ws, size_t ws_size,
                              hipStream_t stream)
{
    (void)in_sizes; (void)n_in; (void)out_size; (void)ws_size;
    const float* h = (const float*)d_in[0];   // (8, 2048, 256) f32
    const float* W = (const float*)d_in[1];   // (256, 128) f32
    const float* a = (const float*)d_in[2];   // (256,) f32
    float* out = (float*)d_out;               // (8, 2048, 128) f32

    // Workspace layout
    __bf16* WhT  = (__bf16*)d_ws;                                   // 4 MB, [b][n][j]
    float*  s1   = (float*)((char*)d_ws + (size_t)Bsz * N * FO * 2);
    float*  s2   = s1 + Bsz * N;
    float*  s2mx = s2 + Bsz * N;

    gat_gemm_scores<<<dim3(Bsz * N / 64), dim3(128), 0, stream>>>(h, W, a, WhT, s1, s2);
    gat_s2max<<<dim3(Bsz), dim3(256), 0, stream>>>(s2, s2mx);
    gat_attention<<<dim3(Bsz * N / 64), dim3(128), 0, stream>>>(WhT, s1, s2, s2mx, out);
}
